// MultiHeadAttention_46368466928108
// MI455X (gfx1250) — matrix operand order repaired and checked by reading, not compiled, not run on hardware
//
#include <hip/hip_runtime.h>
#include <hip/hip_bf16.h>

// ---------------------------------------------------------------------------
// MI455X / gfx1250: bf16 WMMA GEMMs + flash attention.
//   B=2, S=2048, C_IN=1024, H=16, C=64
// fp32->bf16 conversion hoisted into a one-time packed pass so GEMM inner
// loops are pure global_load_b128 + v_wmma. Rotary uses hardware sin/cos
// (v_sin_f32 / v_cos_f32) to keep the epilogue register-lean (no spills).
// ---------------------------------------------------------------------------

typedef __bf16 bf16_t;
typedef __attribute__((ext_vector_type(16))) __bf16 v16bf;
typedef __attribute__((ext_vector_type(8)))  __bf16 v8bf;
typedef __attribute__((ext_vector_type(8)))  float  v8f;
typedef __attribute__((ext_vector_type(4)))  float  v4f;

#define S_LEN 2048
#define BATCH 2
#define NHEAD 16
#define HDIM  64
#define CIN   1024

__device__ __forceinline__ bf16_t f2bf(float f) {
  unsigned u = __builtin_bit_cast(unsigned, f);
  unsigned r = u + 0x7FFFu + ((u >> 16) & 1u);   // round-to-nearest-even
  unsigned short h = (unsigned short)(r >> 16);
  return __builtin_bit_cast(bf16_t, h);
}

// Pack two floats into two bf16 (RNE) in one dword: lo = a, hi = b.
__device__ __forceinline__ unsigned pack2bf(float a, float b) {
  unsigned ua = __builtin_bit_cast(unsigned, a);
  unsigned ub = __builtin_bit_cast(unsigned, b);
  ua = ua + 0x7FFFu + ((ua >> 16) & 1u);
  ub = ub + 0x7FFFu + ((ub >> 16) & 1u);
  return (ua >> 16) | (ub & 0xFFFF0000u);
}

__device__ __forceinline__ v8f wmma_bf16(v16bf a, v16bf b, v8f c) {
  // D = A(16x32 bf16) * B(32x16 bf16) + C(16x16 f32)
  return __builtin_amdgcn_wmma_f32_16x16x32_bf16(
      false, a, false, b, (short)0, c, false, false);
}

// ---------------------------------------------------------------------------
// Kernel 0: one-time fp32 -> packed bf16 conversion (n multiple of 8).
// ---------------------------------------------------------------------------
__global__ __launch_bounds__(256)
void cvt_kernel(const float* __restrict__ src, uint4* __restrict__ dst,
                int n /* floats */)
{
  const int i = blockIdx.x * blockDim.x + threadIdx.x;   // 8 floats / thread
  const int base = i * 8;
  if (base >= n) return;
  v4f a = *(const v4f*)(src + base);
  v4f b = *(const v4f*)(src + base + 4);
  uint4 o;
  o.x = pack2bf(a[0], a[1]);
  o.y = pack2bf(a[2], a[3]);
  o.z = pack2bf(b[0], b[1]);
  o.w = pack2bf(b[2], b[3]);
  dst[i] = o;
}

// ---------------------------------------------------------------------------
// Kernel 1: fused Q/K/V/G projection from pre-converted bf16 operands.
// One wave computes a 32(M) x 64(N) tile (2 M-subtiles x one full head),
// reusing each B fragment for both M-subtiles. blockIdx.z picks the matrix.
// Epilogues: rotary (+1/sqrt(C) for Q), V transposed store, sigmoid gate.
// ---------------------------------------------------------------------------
__global__ __launch_bounds__(128)
void proj_kernel(const bf16_t* __restrict__ xb,
                 const bf16_t* __restrict__ Wqb, const bf16_t* __restrict__ Wkb,
                 const bf16_t* __restrict__ Wvb, const bf16_t* __restrict__ Wgb,
                 const float* __restrict__ bg,
                 bf16_t* __restrict__ Qb, bf16_t* __restrict__ Kb,
                 bf16_t* __restrict__ Vt, float* __restrict__ G)
{
  const int lane = threadIdx.x & 31;
  const int wave = threadIdx.x >> 5;
  const int half = lane >> 4;
  const int l16  = lane & 15;
  const int mt2  = blockIdx.x * 4 + wave;    // 0..127 (32-row blocks, M=4096)
  const int head = blockIdx.y;               // 0..15
  const int mat  = blockIdx.z;               // 0:Q 1:K 2:V 3:G

  const bf16_t* W =
      (mat == 0) ? Wqb : (mat == 1) ? Wkb : (mat == 2) ? Wvb : Wgb;

  const int row0  = mt2 * 32;
  const int nbase = head * HDIM;

  v8f acc[2][4] = {};   // [M-subtile][column group of 16]

  const bf16_t* xrow0 = xb + (size_t)(row0 + l16) * CIN;
  const bf16_t* xrow1 = xb + (size_t)(row0 + 16 + l16) * CIN;

  for (int k0 = 0; k0 < CIN; k0 += 32) {
    const int ksh = k0 + (half ? 16 : 0);
    v16bf af[2];
#pragma unroll
    for (int mi = 0; mi < 2; ++mi) {
      const bf16_t* xr = mi ? xrow1 : xrow0;
      v8bf lo = *(const v8bf*)(xr + ksh);
      v8bf hi = *(const v8bf*)(xr + ksh + 8);
#pragma unroll
      for (int i = 0; i < 8; ++i) { af[mi][i] = lo[i]; af[mi][i + 8] = hi[i]; }
    }
#pragma unroll
    for (int g = 0; g < 4; ++g) {
      // B[k][n] = W[n][k]; column n = lane%16, K contiguous per lane.
      const bf16_t* wrow =
          W + (size_t)(nbase + g * 16 + l16) * CIN + k0 + (half ? 16 : 0);
      v16bf bv = *(const v16bf*)(wrow);
      acc[0][g] = wmma_bf16(af[0], bv, acc[0][g]);
      acc[1][g] = wmma_bf16(af[1], bv, acc[1][g]);
    }
  }

  // Hoisted rotary frequencies: per lane only two distinct j = g*16 + l16.
  // freq = 10000^(-j/32) = exp(-j * ln(10000)/32)
  const float freq0 = __expf(-0.28782313662425572f * (float)(l16));
  const float freq1 = __expf(-0.28782313662425572f * (float)(16 + l16));

  // C/D layout: comp r -> row rowM + r + half*8; column n = g*16 + l16.
#pragma unroll
  for (int mi = 0; mi < 2; ++mi) {
    const int rowM = row0 + mi * 16;
    if (mat <= 1) {
      // Rotary: channel i pairs with i+32 => acc[g] with acc[g+2].
#pragma unroll
      for (int r = 0; r < 8; ++r) {
        const int grow = rowM + r + half * 8;
        const float s = (float)(grow & (S_LEN - 1));
#pragma unroll
        for (int g = 0; g < 2; ++g) {
          const float ang = s * (g ? freq1 : freq0);
          // hw transcendentals: |err| ~2.4e-4 rad at ang~2048, << bf16 eps
          const float c = __cosf(ang), sn = __sinf(ang);
          float x1 = acc[mi][g][r], x2 = acc[mi][g + 2][r];
          float o1 = x1 * c - x2 * sn;
          float o2 = x2 * c + x1 * sn;
          if (mat == 0) { o1 *= 0.125f; o2 *= 0.125f; }     // 1/sqrt(64)
          acc[mi][g][r] = o1; acc[mi][g + 2][r] = o2;
        }
      }
      bf16_t* dst = (mat == 0) ? Qb : Kb;
#pragma unroll
      for (int r = 0; r < 8; ++r) {
        const int grow = rowM + r + half * 8;
        const int b = grow >> 11, s = grow & (S_LEN - 1);
        const size_t base = (((size_t)(b * NHEAD + head)) * S_LEN + s) * HDIM;
#pragma unroll
        for (int g = 0; g < 4; ++g)
          dst[base + g * 16 + l16] = f2bf(acc[mi][g][r]);
      }
    } else if (mat == 2) {
      // V transposed: Vt[(bh*64 + c)*S + s]  (kk contiguous for AV B-frag)
#pragma unroll
      for (int r = 0; r < 8; ++r) {
        const int grow = rowM + r + half * 8;
        const int b = grow >> 11, s = grow & (S_LEN - 1);
        const size_t bh = (size_t)(b * NHEAD + head);
#pragma unroll
        for (int g = 0; g < 4; ++g) {
          const int c = g * 16 + l16;
          Vt[(bh * HDIM + c) * S_LEN + s] = f2bf(acc[mi][g][r]);
        }
      }
    } else {
#pragma unroll
      for (int r = 0; r < 8; ++r) {
        const int grow = rowM + r + half * 8;
#pragma unroll
        for (int g = 0; g < 4; ++g) {
          const int n = nbase + g * 16 + l16;
          const float v = acc[mi][g][r] + bg[n];
          G[(size_t)grow * CIN + n] = 1.0f / (1.0f + __expf(-v));
        }
      }
    }
  }
}

// ---------------------------------------------------------------------------
// Kernel 2: flash attention. One wave per (b, h, 16-row q tile).
// Scores via 2 WMMAs with the bias tile preloaded as the C operand; online
// softmax with half-wave shuffles; P routed through LDS (C-layout ->
// A-layout); AV via 4 WMMAs per 32-wide k step. Gating fused into epilogue.
// ---------------------------------------------------------------------------
__global__ __launch_bounds__(32)
void attn_kernel(const bf16_t* __restrict__ Qb, const bf16_t* __restrict__ Kb,
                 const bf16_t* __restrict__ Vt, const float* __restrict__ bias,
                 const int* __restrict__ amask, const float* __restrict__ G,
                 bf16_t* __restrict__ GO)
{
  __shared__ bf16_t Pl[16 * 32];

  const int lane = threadIdx.x & 31;
  const int half = lane >> 4;
  const int l16  = lane & 15;
  const int qt = blockIdx.x & 127;
  const int h  = (blockIdx.x >> 7) & 15;
  const int b  = blockIdx.x >> 11;
  const size_t bh = (size_t)(b * NHEAD + h);
  const int q0 = qt * 16;

  // Q A-fragments (K-dim = channel): aQ[0] c=0..31, aQ[1] c=32..63
  const bf16_t* qrow = Qb + (bh * S_LEN + q0 + l16) * HDIM;
  v16bf aQ[2];
#pragma unroll
  for (int t = 0; t < 2; ++t) {
    const int c0 = t * 32 + (half ? 16 : 0);
    v8bf lo = *(const v8bf*)(qrow + c0);
    v8bf hi = *(const v8bf*)(qrow + c0 + 8);
#pragma unroll
    for (int i = 0; i < 8; ++i) { aQ[t][i] = lo[i]; aQ[t][i + 8] = hi[i]; }
  }

  float mrow[8], lrow[8];
#pragma unroll
  for (int r = 0; r < 8; ++r) { mrow[r] = -1e30f; lrow[r] = 0.0f; }
  v8f oacc[4] = {};

  const float* biasrow = bias + (bh * S_LEN + q0) * (size_t)S_LEN;

  for (int k0 = 0; k0 < S_LEN; k0 += 32) {
    if (k0 + 32 < S_LEN)   // speculative L2 prefetch of next bias block
      __builtin_prefetch(biasrow + (size_t)l16 * S_LEN + k0 + 32, 0, 1);

    v8f sc[2];
#pragma unroll
    for (int kt = 0; kt < 2; ++kt) {
      const int kb = k0 + kt * 16;
      v8f cfr;                              // bias tile as WMMA C operand
#pragma unroll
      for (int r = 0; r < 8; ++r)
        cfr[r] = biasrow[(size_t)(r + half * 8) * S_LEN + kb + l16];
      // K^T B-fragments: column kk = lane%16, K-dim = channel contiguous
      const bf16_t* krow = Kb + (bh * S_LEN + kb + l16) * HDIM;
      v16bf bK0 = *(const v16bf*)(krow + (half ? 16 : 0));
      v16bf bK1 = *(const v16bf*)(krow + 32 + (half ? 16 : 0));
      v8f s = wmma_bf16(aQ[0], bK0, cfr);
      s = wmma_bf16(aQ[1], bK1, s);
      const float mk = (amask[b * S_LEN + kb + l16] == 0) ? -1e8f : 0.0f;
#pragma unroll
      for (int r = 0; r < 8; ++r) s[r] += mk;
      sc[kt] = s;
    }

    // Online softmax: row stats live per (comp r, 16-lane half).
#pragma unroll
    for (int r = 0; r < 8; ++r) {
      float v = fmaxf(sc[0][r], sc[1][r]);
      v = fmaxf(v, __shfl_xor(v, 1, 32));
      v = fmaxf(v, __shfl_xor(v, 2, 32));
      v = fmaxf(v, __shfl_xor(v, 4, 32));
      v = fmaxf(v, __shfl_xor(v, 8, 32));
      const float mn = fmaxf(mrow[r], v);
      const float scale = __expf(mrow[r] - mn);
      mrow[r] = mn;
      const float p0 = __expf(sc[0][r] - mn);
      const float p1 = __expf(sc[1][r] - mn);
      sc[0][r] = p0; sc[1][r] = p1;
      float rs = p0 + p1;
      rs += __shfl_xor(rs, 1, 32);
      rs += __shfl_xor(rs, 2, 32);
      rs += __shfl_xor(rs, 4, 32);
      rs += __shfl_xor(rs, 8, 32);
      lrow[r] = lrow[r] * scale + rs;
#pragma unroll
      for (int g = 0; g < 4; ++g) oacc[g][r] *= scale;
    }

    // P: C-layout -> LDS [m][kk] -> A-layout bf16 fragment.
    asm volatile("s_wait_dscnt 0" ::: "memory");    // WAR vs prev-iter reads
#pragma unroll
    for (int kt = 0; kt < 2; ++kt)
#pragma unroll
      for (int r = 0; r < 8; ++r)
        Pl[(r + half * 8) * 32 + kt * 16 + l16] = f2bf(sc[kt][r]);
    asm volatile("s_wait_dscnt 0" ::: "memory");    // RAW before A-frag reads

    v16bf pA;
    {
      const bf16_t* prow = Pl + l16 * 32 + (half ? 16 : 0);
      v8bf lo = *(const v8bf*)(prow);
      v8bf hi = *(const v8bf*)(prow + 8);
#pragma unroll
      for (int i = 0; i < 8; ++i) { pA[i] = lo[i]; pA[i + 8] = hi[i]; }
    }

    // AV: B[kk][c] from transposed V (kk contiguous per lane).
#pragma unroll
    for (int g = 0; g < 4; ++g) {
      const bf16_t* vrow =
          Vt + (bh * HDIM + g * 16 + l16) * S_LEN + k0 + (half ? 16 : 0);
      v16bf bV = *(const v16bf*)(vrow);
      oacc[g] = wmma_bf16(pA, bV, oacc[g]);
    }
  }

  // Normalize, apply gate, store (B*S, H*C) bf16 for the output projection.
#pragma unroll
  for (int r = 0; r < 8; ++r) {
    const int grow = b * S_LEN + q0 + r + half * 8;
    const float inv = 1.0f / lrow[r];
#pragma unroll
    for (int g = 0; g < 4; ++g) {
      const int n = h * HDIM + g * 16 + l16;
      const float o = oacc[g][r] * inv;
      const float gt = G[(size_t)grow * CIN + n];
      GO[(size_t)grow * CIN + n] = f2bf(o * gt);
    }
  }
}

// ---------------------------------------------------------------------------
// Kernel 3: out = (g*o) @ Wo^T + bo, fp32 output. 32x64 tile per wave.
// ---------------------------------------------------------------------------
__global__ __launch_bounds__(128)
void outproj_kernel(const bf16_t* __restrict__ GO,
                    const bf16_t* __restrict__ Wob,
                    const float* __restrict__ bo, float* __restrict__ out)
{
  const int lane = threadIdx.x & 31;
  const int wave = threadIdx.x >> 5;
  const int half = lane >> 4;
  const int l16  = lane & 15;
  const int mt2   = blockIdx.x * 4 + wave;  // 0..127
  const int ntile = blockIdx.y;             // 0..15 -> 64 output channels
  const int row0 = mt2 * 32, nbase = ntile * 64;

  v8f acc[2][4] = {};
  const bf16_t* arow0 = GO + (size_t)(row0 + l16) * CIN;
  const bf16_t* arow1 = GO + (size_t)(row0 + 16 + l16) * CIN;

  for (int k0 = 0; k0 < CIN; k0 += 32) {
    const int ksh = k0 + (half ? 16 : 0);
    v16bf af[2];
#pragma unroll
    for (int mi = 0; mi < 2; ++mi) {
      const bf16_t* ar = mi ? arow1 : arow0;
      v8bf lo = *(const v8bf*)(ar + ksh);
      v8bf hi = *(const v8bf*)(ar + ksh + 8);
#pragma unroll
      for (int i = 0; i < 8; ++i) { af[mi][i] = lo[i]; af[mi][i + 8] = hi[i]; }
    }
#pragma unroll
    for (int g = 0; g < 4; ++g) {
      const bf16_t* wrow =
          Wob + (size_t)(nbase + g * 16 + l16) * CIN + k0 + (half ? 16 : 0);
      v16bf bv = *(const v16bf*)(wrow);
      acc[0][g] = wmma_bf16(af[0], bv, acc[0][g]);
      acc[1][g] = wmma_bf16(af[1], bv, acc[1][g]);
    }
  }
#pragma unroll
  for (int mi = 0; mi < 2; ++mi) {
#pragma unroll
    for (int r = 0; r < 8; ++r) {
      const int grow = row0 + mi * 16 + r + half * 8;
#pragma unroll
      for (int g = 0; g < 4; ++g) {
        const int n = nbase + g * 16 + l16;
        out[(size_t)grow * CIN + n] = acc[mi][g][r] + bo[n];
      }
    }
  }
}

// ---------------------------------------------------------------------------
extern "C" void kernel_launch(void* const* d_in, const int* in_sizes, int n_in,
                              void* d_out, int out_size, void* d_ws, size_t ws_size,
                              hipStream_t stream)
{
  const float* x    = (const float*)d_in[0];
  const float* bias = (const float*)d_in[1];
  const float* Wq   = (const float*)d_in[2];
  const float* Wk   = (const float*)d_in[3];
  const float* Wv   = (const float*)d_in[4];
  const float* Wg   = (const float*)d_in[5];
  const float* bg   = (const float*)d_in[6];
  const float* Wo   = (const float*)d_in[7];
  const float* bo   = (const float*)d_in[8];
  const int*   am   = (const int*)d_in[9];
  float* out = (float*)d_out;

  // Workspace layout (66 MB total):
  char* ws = (char*)d_ws;
  bf16_t* xb  = (bf16_t*)ws;  ws += (size_t)8 << 20;   // x, bf16
  bf16_t* Wqb = (bf16_t*)ws;  ws += (size_t)2 << 20;
  bf16_t* Wkb = (bf16_t*)ws;  ws += (size_t)2 << 20;
  bf16_t* Wvb = (bf16_t*)ws;  ws += (size_t)2 << 20;
  bf16_t* Wgb = (bf16_t*)ws;  ws += (size_t)2 << 20;
  bf16_t* Wob = (bf16_t*)ws;  ws += (size_t)2 << 20;
  bf16_t* Qb  = (bf16_t*)ws;  ws += (size_t)8 << 20;   // rope(q)/8, bf16
  bf16_t* Kb  = (bf16_t*)ws;  ws += (size_t)8 << 20;   // rope(k), bf16
  bf16_t* Vt  = (bf16_t*)ws;  ws += (size_t)8 << 20;   // V transposed, bf16
  float*  G   = (float*)ws;   ws += (size_t)16 << 20;  // sigmoid gate, f32
  bf16_t* GO  = (bf16_t*)ws;                           // g*o, bf16

  const int NX = BATCH * S_LEN * CIN;   // 4M floats
  const int NW = CIN * CIN;             // 1M floats per weight
  cvt_kernel<<<dim3(NX / 8 / 256), 256, 0, stream>>>(x,  (uint4*)xb,  NX);
  cvt_kernel<<<dim3(NW / 8 / 256), 256, 0, stream>>>(Wq, (uint4*)Wqb, NW);
  cvt_kernel<<<dim3(NW / 8 / 256), 256, 0, stream>>>(Wk, (uint4*)Wkb, NW);
  cvt_kernel<<<dim3(NW / 8 / 256), 256, 0, stream>>>(Wv, (uint4*)Wvb, NW);
  cvt_kernel<<<dim3(NW / 8 / 256), 256, 0, stream>>>(Wg, (uint4*)Wgb, NW);
  cvt_kernel<<<dim3(NW / 8 / 256), 256, 0, stream>>>(Wo, (uint4*)Wob, NW);

  proj_kernel<<<dim3(32, 16, 4), 128, 0, stream>>>(xb, Wqb, Wkb, Wvb, Wgb, bg,
                                                   Qb, Kb, Vt, G);
  attn_kernel<<<dim3(BATCH * NHEAD * (S_LEN / 16)), 32, 0, stream>>>(
      Qb, Kb, Vt, bias, am, G, GO);
  outproj_kernel<<<dim3(32, 16), 128, 0, stream>>>(GO, Wob, bo, out);
}
